// RecurrentCellsModel_42219528520302
// MI455X (gfx1250) — matrix operand order repaired and checked by reading, not compiled, not run on hardware
//
#include <hip/hip_runtime.h>

// Problem constants (match reference)
#define B_  128
#define T_  2048
#define D_  256
#define H_  512
#define O_  6

typedef __attribute__((ext_vector_type(16))) __bf16 v16bf;
typedef __attribute__((ext_vector_type(8)))  float  v8f;

static __device__ __forceinline__ float sigmoidf_(float x) {
  return 1.0f / (1.0f + __expf(-x));
}

static __device__ __forceinline__ v8f wmma_bf16(v16bf a, v16bf b, v8f c) {
  // (neg_a, A, neg_b, B, c_mod, C, reuse_a, reuse_b)
  return __builtin_amdgcn_wmma_f32_16x16x32_bf16(false, a, false, b, (short)0, c,
                                                 false, false);
}

// Opaque zero (uniform, in an SGPR).  Volatile asm cannot be hoisted or
// CSE'd, so adding this to a block index inside the time loop makes all
// weight-fragment addresses loop-variant (prevents LICM from preloading ~200
// fragments and spilling them to scratch) WITHOUT obscuring the base pointer
// -- address-space inference still sees a global pointer and emits
// global_load_b128 (LOADcnt only) instead of flat_load (LOADcnt+DScnt).
static __device__ __forceinline__ int opaque_zero() {
  int x = 0;
  asm volatile("" : "+s"(x));
  return x;
}

// ---------------------------------------------------------------------------
// A-fragment (16x32 bf16) from LDS, row-major [16][Kdim].
// ISA 16-bit A layout: lanes 0-15 hold row M=lane, K = {ks..ks+7, ks+16..ks+23};
// lanes 16-31 hold row M=lane-16, K = {ks+8..ks+15, ks+24..ks+31}.
// ---------------------------------------------------------------------------
static __device__ __forceinline__ v16bf load_a(const __bf16* lds, int Kdim,
                                               int lane, int ks) {
  const int row = lane & 15;
  const int kb  = ks + ((lane >> 4) << 3);
  const __bf16* p = lds + row * Kdim + kb;
  v16bf a;
#pragma unroll
  for (int i = 0; i < 8; ++i) a[i] = p[i];          // K = kb .. kb+7
#pragma unroll
  for (int i = 0; i < 8; ++i) a[8 + i] = p[16 + i]; // K = kb+16 .. kb+23
  return a;
}

// ---------------------------------------------------------------------------
// B-fragment (32x16 bf16) from global, pre-packed fragment-major by wconv():
// one block = 512 halves = 1 KB; lane l reads 16 contiguous halves (32 B).
// ---------------------------------------------------------------------------
static __device__ __forceinline__ v16bf load_b(const __bf16* w, int block, int lane) {
  return *(const v16bf*)(w + (size_t)block * 512 + lane * 16);
}

// ---------------------------------------------------------------------------
// Pack a row-major fp32 weight W[N][K] (lin = x @ W.T) into bf16 B-fragments.
// B(k,n) = W[n][k]; block = (n/16)*(K/32) + (k/32);
// lane = (n%16) + 16*((k>>3)&1); sub = (k & 7) + 8*((k>>4)&1).
// ---------------------------------------------------------------------------
__global__ void wconv(const float* __restrict__ w, __bf16* __restrict__ dst,
                      int N, int K) {
  int idx = blockIdx.x * blockDim.x + threadIdx.x;
  if (idx >= N * K) return;
  int n = idx / K;
  int k = idx - n * K;
  int block = (n >> 4) * (K >> 5) + (k >> 5);
  int lane  = (n & 15) + ((k & 8) << 1);
  int sub   = (k & 7) + ((k & 16) >> 1);
  dst[(size_t)block * 512 + lane * 16 + sub] = (__bf16)w[idx];
}

// ---------------------------------------------------------------------------
// Persistent recurrent kernel: one workgroup owns 16 batch rows for all T
// steps.  32 waves; wave w owns output columns [w*16, w*16+16) of H, and the
// matching column tiles of every gate block, so gate accumulators are
// element-aligned and the nonlinearities fuse in registers.
// ---------------------------------------------------------------------------
__global__ __launch_bounds__(1024, 1)
void recurrent_kernel(const int* __restrict__ tok, const float* __restrict__ emb,
                      const __bf16* __restrict__ wihB,   // GRU Wih  (1536 x 256)
                      const __bf16* __restrict__ whhB,   // GRU Whh  (1536 x 512)
                      const __bf16* __restrict__ lwihB,  // LSTM Wih (2048 x 512)
                      const __bf16* __restrict__ lwhhB,  // LSTM Whh (2048 x 512)
                      const float* __restrict__ gru_b,
                      const float* __restrict__ gru_bn,
                      const float* __restrict__ lstm_b,
                      float* __restrict__ hfinal) {
  __shared__ __bf16 xtA[16 * D_];        //  8 KB  staged embeddings (A matrix)
  __shared__ __bf16 hgA[2][16 * H_];     // 32 KB  GRU hidden, double buffered
  __shared__ __bf16 hlA[2][16 * H_];     // 32 KB  LSTM hidden, double buffered
  __shared__ float  cF[16 * H_];         // 32 KB  LSTM cell state (fp32)

  const int tid  = threadIdx.x;
  const int lane = tid & 31;
  const int w    = tid >> 5;             // wave id 0..31
  const int b0   = blockIdx.x * 16;      // batch tile base

  // zero-init carried state (h0 == 0)
  for (int i = tid; i < 16 * H_; i += 1024) {
    hgA[0][i] = (__bf16)0.0f;
    hlA[0][i] = (__bf16)0.0f;
    cF[i]     = 0.0f;
  }
  __syncthreads();

  const int n0  = w << 4;                // column tile base in [0,H)
  const int col = n0 + (lane & 15);      // this lane's output column
  const int Mofs = (lane >> 4) << 3;     // C-layout row offset (0 or 8)

  // per-lane bias constants (same column for all 8 C registers)
  const float bR = gru_b[0 * H_ + col];
  const float bZ = gru_b[1 * H_ + col];
  const float bN = gru_b[2 * H_ + col];
  const float bn2 = gru_bn[col];
  const float bI = lstm_b[0 * H_ + col];
  const float bF = lstm_b[1 * H_ + col];
  const float bG = lstm_b[2 * H_ + col];
  const float bO = lstm_b[3 * H_ + col];

  for (int t = 0; t < T_; ++t) {
    const int cur = t & 1, nxt = cur ^ 1;

    // ---- stage xt = emb[token_ids[b0..b0+15, t]] as bf16 into LDS ----
    // (token load depends on t, so these are naturally loop-variant:
    //  no laundering needed, plain global loads)
    {
      const int flat = tid << 2;                 // 4 elems / thread
      const int r = flat >> 8;                   // row 0..15
      const int c = flat & (D_ - 1);
      const int token = tok[(b0 + r) * T_ + t];
      const float* er = emb + (size_t)token * D_ + c;
#pragma unroll
      for (int i = 0; i < 4; ++i) xtA[r * D_ + c + i] = (__bf16)er[i];
    }
    __syncthreads();

    // =================== GRU cell ===================
    v8f aR  = {bR, bR, bR, bR, bR, bR, bR, bR};
    v8f aZ  = {bZ, bZ, bZ, bZ, bZ, bZ, bZ, bZ};
    v8f aNi = {bN, bN, bN, bN, bN, bN, bN, bN};
    v8f aNh = {0.f, 0.f, 0.f, 0.f, 0.f, 0.f, 0.f, 0.f};

    // ig = xt @ Wih.T  (K = 256, folded into the step: no big intermediate)
    {
      const int zb = opaque_zero();
#pragma unroll
      for (int kt = 0; kt < D_ / 32; ++kt) {
        v16bf a = load_a(xtA, D_, lane, kt * 32);
        aR  = wmma_bf16(a, load_b(wihB, zb + (0 * 32 + w) * 8 + kt, lane), aR);
        aZ  = wmma_bf16(a, load_b(wihB, zb + (1 * 32 + w) * 8 + kt, lane), aZ);
        aNi = wmma_bf16(a, load_b(wihB, zb + (2 * 32 + w) * 8 + kt, lane), aNi);
      }
    }
    // hh = hg @ Whh.T  (K = 512)
    {
      const int zb = opaque_zero();
#pragma unroll
      for (int kt = 0; kt < H_ / 32; ++kt) {
        v16bf a = load_a(hgA[cur], H_, lane, kt * 32);
        if (kt < 15)
          __builtin_prefetch(whhB + (size_t)(zb + (0 * 32 + w) * 16 + kt + 1) * 512 +
                                 lane * 16, 0, 1);
        aR  = wmma_bf16(a, load_b(whhB, zb + (0 * 32 + w) * 16 + kt, lane), aR);
        aZ  = wmma_bf16(a, load_b(whhB, zb + (1 * 32 + w) * 16 + kt, lane), aZ);
        aNh = wmma_bf16(a, load_b(whhB, zb + (2 * 32 + w) * 16 + kt, lane), aNh);
      }
    }
    // fused gate nonlinearities (accumulators element-aligned in C layout)
#pragma unroll
    for (int r = 0; r < 8; ++r) {
      const int row = r + Mofs;
      float rr = sigmoidf_(aR[r]);
      float zz = sigmoidf_(aZ[r]);
      float nn = tanhf(aNi[r] + rr * (aNh[r] + bn2));
      float hp = (float)hgA[cur][row * H_ + col];
      float hn = nn + zz * (hp - nn);
      hgA[nxt][row * H_ + col] = (__bf16)hn;
    }
    __syncthreads();

    // =================== LSTM cell ===================
    v8f aI = {bI, bI, bI, bI, bI, bI, bI, bI};
    v8f aF = {bF, bF, bF, bF, bF, bF, bF, bF};
    v8f aG = {bG, bG, bG, bG, bG, bG, bG, bG};
    v8f aO = {bO, bO, bO, bO, bO, bO, bO, bO};

    // hg_new @ lstm_wih.T
    {
      const int zb = opaque_zero();
#pragma unroll
      for (int kt = 0; kt < H_ / 32; ++kt) {
        v16bf a = load_a(hgA[nxt], H_, lane, kt * 32);
        aI = wmma_bf16(a, load_b(lwihB, zb + (0 * 32 + w) * 16 + kt, lane), aI);
        aF = wmma_bf16(a, load_b(lwihB, zb + (1 * 32 + w) * 16 + kt, lane), aF);
        aG = wmma_bf16(a, load_b(lwihB, zb + (2 * 32 + w) * 16 + kt, lane), aG);
        aO = wmma_bf16(a, load_b(lwihB, zb + (3 * 32 + w) * 16 + kt, lane), aO);
      }
    }
    // hl @ lstm_whh.T
    {
      const int zb = opaque_zero();
#pragma unroll
      for (int kt = 0; kt < H_ / 32; ++kt) {
        v16bf a = load_a(hlA[cur], H_, lane, kt * 32);
        if (kt < 15)
          __builtin_prefetch(lwhhB + (size_t)(zb + (0 * 32 + w) * 16 + kt + 1) * 512 +
                                 lane * 16, 0, 1);
        aI = wmma_bf16(a, load_b(lwhhB, zb + (0 * 32 + w) * 16 + kt, lane), aI);
        aF = wmma_bf16(a, load_b(lwhhB, zb + (1 * 32 + w) * 16 + kt, lane), aF);
        aG = wmma_bf16(a, load_b(lwhhB, zb + (2 * 32 + w) * 16 + kt, lane), aG);
        aO = wmma_bf16(a, load_b(lwhhB, zb + (3 * 32 + w) * 16 + kt, lane), aO);
      }
    }
#pragma unroll
    for (int r = 0; r < 8; ++r) {
      const int row = r + Mofs;
      float ii = sigmoidf_(aI[r]);
      float ff = sigmoidf_(aF[r]);
      float gg = tanhf(aG[r]);
      float oo = sigmoidf_(aO[r]);
      float c_ = ff * cF[row * H_ + col] + ii * gg;  // each element lane-owned
      cF[row * H_ + col] = c_;
      float h_ = oo * tanhf(c_);
      hlA[nxt][row * H_ + col] = (__bf16)h_;
      if (t == T_ - 1) hfinal[(size_t)(b0 + row) * H_ + col] = h_;
    }
    __syncthreads();
  }
}

// ---------------------------------------------------------------------------
// Head: out = h_final @ head_w.T + head_b   (128x512 · 512x6 — tiny)
// ---------------------------------------------------------------------------
__global__ void head_kernel(const float* __restrict__ hfinal,
                            const float* __restrict__ head_w,
                            const float* __restrict__ head_b,
                            float* __restrict__ out) {
  int idx = blockIdx.x * blockDim.x + threadIdx.x;
  if (idx >= B_ * O_) return;
  int b = idx / O_;
  int o = idx - b * O_;
  const float* h  = hfinal + (size_t)b * H_;
  const float* wr = head_w + (size_t)o * H_;
  float s = 0.0f;
#pragma unroll 8
  for (int k = 0; k < H_; ++k) s = fmaf(h[k], wr[k], s);
  out[idx] = s + head_b[o];
}

extern "C" void kernel_launch(void* const* d_in, const int* in_sizes, int n_in,
                              void* d_out, int out_size, void* d_ws, size_t ws_size,
                              hipStream_t stream) {
  (void)in_sizes; (void)n_in; (void)out_size; (void)ws_size;

  const int*   tok      = (const int*)d_in[0];
  const float* emb      = (const float*)d_in[1];
  const float* gru_wih  = (const float*)d_in[2];   // (3H, D) = 1536 x 256
  const float* gru_whh  = (const float*)d_in[3];   // (3H, H) = 1536 x 512
  const float* gru_b    = (const float*)d_in[4];
  const float* gru_bn   = (const float*)d_in[5];
  const float* lstm_wih = (const float*)d_in[6];   // (4H, H) = 2048 x 512
  const float* lstm_whh = (const float*)d_in[7];   // (4H, H) = 2048 x 512
  const float* lstm_b   = (const float*)d_in[8];
  const float* head_w   = (const float*)d_in[9];
  const float* head_b   = (const float*)d_in[10];
  float*       out      = (float*)d_out;

  // workspace layout (bytes)
  char* ws = (char*)d_ws;
  size_t off = 0;
  __bf16* wihB  = (__bf16*)(ws + off); off += (size_t)(3 * H_) * D_ * 2;  // 768 KB
  __bf16* whhB  = (__bf16*)(ws + off); off += (size_t)(3 * H_) * H_ * 2;  // 1.5 MB
  __bf16* lwihB = (__bf16*)(ws + off); off += (size_t)(4 * H_) * H_ * 2;  // 2 MB
  __bf16* lwhhB = (__bf16*)(ws + off); off += (size_t)(4 * H_) * H_ * 2;  // 2 MB
  float*  hfin  = (float*)(ws + off);  off += (size_t)B_ * H_ * 4;        // 256 KB

  // 1) pack weights into bf16 WMMA B-fragments (L2-resident thereafter)
  {
    int n;
    n = (3 * H_) * D_; wconv<<<(n + 255) / 256, 256, 0, stream>>>(gru_wih,  wihB,  3 * H_, D_);
    n = (3 * H_) * H_; wconv<<<(n + 255) / 256, 256, 0, stream>>>(gru_whh,  whhB,  3 * H_, H_);
    n = (4 * H_) * H_; wconv<<<(n + 255) / 256, 256, 0, stream>>>(lstm_wih, lwihB, 4 * H_, H_);
    n = (4 * H_) * H_; wconv<<<(n + 255) / 256, 256, 0, stream>>>(lstm_whh, lwhhB, 4 * H_, H_);
  }

  // 2) persistent batch-decomposed recurrence: 8 WGs x 1024 threads,
  //    zero grid-level synchronization.
  recurrent_kernel<<<B_ / 16, 1024, 0, stream>>>(
      tok, emb, wihB, whhB, lwihB, lwhhB, gru_b, gru_bn, lstm_b, hfin);

  // 3) tiny output head
  head_kernel<<<(B_ * O_ + 255) / 256, 256, 0, stream>>>(hfin, head_w, head_b, out);
}
